// GCNModel_for_NC_90984587199045
// MI455X (gfx1250) — matrix operand order repaired and checked by reading, not compiled, not run on hardware
//
#include <hip/hip_runtime.h>
#include <hip/hip_bf16.h>

// ---------------------------------------------------------------------------
// CDNA5 (gfx1250) GCN pipeline: bf16 WMMA GEMMs + hardware FP32 L2 atomics
// ---------------------------------------------------------------------------

typedef __attribute__((ext_vector_type(16))) __bf16 v16bf;
typedef __attribute__((ext_vector_type(8)))  float  v8f;

#define GCN_N 50000
#define GCN_E 800000

// Load 16 bf16 values (two 16-byte chunks) into a WMMA fragment register set.
static __device__ __forceinline__ v16bf load_frag16(const __bf16* lo, const __bf16* hi) {
    union { v16bf v; uint4 u[2]; } r;
    r.u[0] = *reinterpret_cast<const uint4*>(lo);
    r.u[1] = *reinterpret_cast<const uint4*>(hi);
    return r.v;
}

// ---------------------------------------------------------------------------
// GEMM: out[M,Nc] = (A[M,K] @ W[K,Nc]) * norm[row]
// A: bf16 row-major.  Wt: bf16 [Nc,K] (transposed weights).
// Block = 128 threads = 4 waves; wave w computes TN 16x16 tiles at
// (m0, blockIdx.y*64*TN + 16*TN*w + 16*t), sharing one A fragment across the
// TN tiles (halves A traffic for TN=2).  K stepped by 32 per
// v_wmma_f32_16x16x32_bf16.
// ---------------------------------------------------------------------------
template<int TN>
__global__ void __launch_bounds__(128)
gemm_bf16_norm(const __bf16* __restrict__ A,
               const __bf16* __restrict__ Wt,
               const float*  __restrict__ norm,
               float* __restrict__ out,
               int M, int K, int Nc)
{
    const int wave = threadIdx.x >> 5;
    const int lane = threadIdx.x & 31;
    const int half = lane >> 4;        // lane group 0..15 vs 16..31
    const int l16  = lane & 15;
    const int m0   = blockIdx.x << 4;
    const int n0   = blockIdx.y * (64 * TN) + wave * (16 * TN);

    // A fragment: lanes 0-15 -> row m0+l16, K = {k..k+7, k+16..k+23}
    //             lanes 16-31 -> same row,  K = {k+8..k+15, k+24..k+31}
    int arow = m0 + l16; if (arow >= M) arow = M - 1;
    const __bf16* aptr = A + (size_t)arow * K + (half << 3);
    // B fragment: lane -> column; K = {k+8*half..+7, k+16+8*half..+7} in Wt
    const __bf16* bptr[TN];
    #pragma unroll
    for (int t = 0; t < TN; ++t)
        bptr[t] = Wt + (size_t)(n0 + 16 * t + l16) * K + (half << 3);

    v8f c[TN];
    #pragma unroll
    for (int t = 0; t < TN; ++t) c[t] = (v8f){};

    for (int k = 0; k < K; k += 32) {
        v16bf a = load_frag16(aptr + k, aptr + k + 16);
        __builtin_prefetch(aptr + k + 32, 0, 3);   // global_prefetch_b8 (near)
        #pragma unroll
        for (int t = 0; t < TN; ++t) {
            v16bf b = load_frag16(bptr[t] + k, bptr[t] + k + 16);
            c[t] = __builtin_amdgcn_wmma_f32_16x16x32_bf16(
                       /*neg_a=*/false, a, /*neg_b=*/false, b,
                       /*c_mod=*/(short)0, c[t],
                       /*reuse_a=*/false, /*reuse_b=*/false);
        }
    }

    // C/D layout: lane group selects M block (0..7 vs 8..15), VGPR r = row.
    const int rbase = m0 + (half << 3);
    #pragma unroll
    for (int t = 0; t < TN; ++t) {
        const int col = n0 + 16 * t + l16;
        #pragma unroll
        for (int r = 0; r < 8; ++r) {
            int row = rbase + r;
            if (row < M && col < Nc)
                out[(size_t)row * Nc + col] = c[t][r] * norm[row];
        }
    }
}

// ---------------------------------------------------------------------------
// Edge aggregation: agg[dst[e]][f] += pre[src[e]][f]  (hardware f32 atomics)
// One thread per float4: 128-bit gather + 4 atomics, indices loaded once.
// ---------------------------------------------------------------------------
template<int LOGC>
__global__ void edge_scatter_add4(const float* __restrict__ pre,
                                  const int* __restrict__ src,
                                  const int* __restrict__ dst,
                                  float* __restrict__ agg, int nEdges)
{
    const int C   = 1 << LOGC;
    const int CV  = C >> 2;            // float4 chunks per row
    long long idx = (long long)blockIdx.x * blockDim.x + threadIdx.x;
    if (idx >= (long long)nEdges * CV) return;
    int e  = (int)(idx >> (LOGC - 2));
    int f4 = (int)idx & (CV - 1);
    const float4 v = reinterpret_cast<const float4*>(pre + (size_t)src[e] * C)[f4];
    float* d = agg + (size_t)dst[e] * C + (f4 << 2);
    unsafeAtomicAdd(d + 0, v.x);       // global_atomic_add_f32
    unsafeAtomicAdd(d + 1, v.y);
    unsafeAtomicAdd(d + 2, v.z);
    unsafeAtomicAdd(d + 3, v.w);
}

// ---------------------------------------------------------------------------
// PairNorm stats: h = agg*norm + b; per-row L2 norm (wave shuffle reduce);
// column sums accumulated in registers then one atomic per (wave, column).
// One wave per row, lane covers columns lane, lane+32, ...
// ---------------------------------------------------------------------------
template<int C>
__global__ void __launch_bounds__(256)
post_stats(const float* __restrict__ agg,
           const float* __restrict__ norm,
           const float* __restrict__ bias,
           float* __restrict__ h,
           float* __restrict__ rownorm,
           float* __restrict__ colsum,
           int M)
{
    constexpr int FPL = C / 32;
    const int lane   = threadIdx.x & 31;
    const int waveId = blockIdx.x * (blockDim.x >> 5) + (threadIdx.x >> 5);
    const int nWaves = gridDim.x * (blockDim.x >> 5);

    float bl[FPL], cs[FPL];
    #pragma unroll
    for (int j = 0; j < FPL; ++j) { bl[j] = bias[lane + 32 * j]; cs[j] = 0.f; }

    for (int row = waveId; row < M; row += nWaves) {
        float nv = norm[row];
        float v[FPL], ss = 0.f;
        #pragma unroll
        for (int j = 0; j < FPL; ++j) {
            v[j] = agg[(size_t)row * C + lane + 32 * j] * nv + bl[j];
            ss += v[j] * v[j];
        }
        #pragma unroll
        for (int o = 16; o > 0; o >>= 1) ss += __shfl_xor(ss, o, 32);
        if (lane == 0) rownorm[row] = sqrtf(1e-6f + ss);
        #pragma unroll
        for (int j = 0; j < FPL; ++j) {
            h[(size_t)row * C + lane + 32 * j] = v[j];
            cs[j] += v[j];
        }
    }
    #pragma unroll
    for (int j = 0; j < FPL; ++j) unsafeAtomicAdd(&colsum[lane + 32 * j], cs[j]);
}

// ---------------------------------------------------------------------------
// Finalize PairNorm + ReLU, emit bf16 activations for the next WMMA GEMM.
// ---------------------------------------------------------------------------
template<int LOGC>
__global__ void finalize_relu_bf16(const float* __restrict__ h,
                                   const float* __restrict__ rownorm,
                                   const float* __restrict__ colsum,
                                   __hip_bfloat16* __restrict__ outbf,
                                   int M)
{
    const int C = 1 << LOGC;
    long long idx = (long long)blockIdx.x * blockDim.x + threadIdx.x;
    if (idx >= ((long long)M << LOGC)) return;
    int row = (int)(idx >> LOGC);
    int f   = (int)idx & (C - 1);
    float v = h[idx] / rownorm[row] - colsum[f] * (1.0f / (float)M);
    v = fmaxf(v, 0.f);
    outbf[idx] = __float2bfloat16(v);
}

// Last layer: out = agg*norm + b  (fp32, no PairNorm/ReLU)
template<int LOGC>
__global__ void final_layer(const float* __restrict__ agg,
                            const float* __restrict__ norm,
                            const float* __restrict__ bias,
                            float* __restrict__ out, int M)
{
    const int C = 1 << LOGC;
    long long idx = (long long)blockIdx.x * blockDim.x + threadIdx.x;
    if (idx >= ((long long)M << LOGC)) return;
    int row = (int)(idx >> LOGC);
    int f   = (int)idx & (C - 1);
    out[idx] = agg[idx] * norm[row] + bias[f];
}

// ---------------------------------------------------------------------------
// Utility kernels
// ---------------------------------------------------------------------------
__global__ void f32_to_bf16(const float* __restrict__ x,
                            __hip_bfloat16* __restrict__ y, long long n)
{
    long long i = (long long)blockIdx.x * blockDim.x + threadIdx.x;
    if (i < n) y[i] = __float2bfloat16(x[i]);
}

// Wt[n*K + k] = bf16(W[k*Nc + n])   (transpose + convert; W is tiny, L2-hot)
__global__ void w_transpose_bf16(const float* __restrict__ W,
                                 __hip_bfloat16* __restrict__ Wt, int K, int Nc)
{
    int i = blockIdx.x * blockDim.x + threadIdx.x;
    if (i >= K * Nc) return;
    int kk = i / Nc, n = i - kk * Nc;
    Wt[(size_t)n * K + kk] = __float2bfloat16(W[i]);
}

__global__ void zero_f32(float* __restrict__ p, long long n)
{
    long long i = (long long)blockIdx.x * blockDim.x + threadIdx.x;
    if (i < n) p[i] = 0.f;
}

// ---------------------------------------------------------------------------
// Host launch
// ---------------------------------------------------------------------------
extern "C" void kernel_launch(void* const* d_in, const int* in_sizes, int n_in,
                              void* d_out, int out_size, void* d_ws, size_t ws_size,
                              hipStream_t stream)
{
    const float* in_feat = (const float*)d_in[0];
    const int*   src     = (const int*)  d_in[1];
    const int*   dst     = (const int*)  d_in[2];
    const float* norm    = (const float*)d_in[3];
    const float* W1      = (const float*)d_in[4];
    const float* b1      = (const float*)d_in[5];
    const float* W2      = (const float*)d_in[6];
    const float* b2      = (const float*)d_in[7];
    const float* W3      = (const float*)d_in[8];
    const float* b3      = (const float*)d_in[9];
    float* out = (float*)d_out;

    const int M = GCN_N;
    const int E = in_sizes[1];

    // Workspace layout (all offsets 64B-aligned)
    char* ws = (char*)d_ws;
    __hip_bfloat16* ABF = (__hip_bfloat16*)(ws);                        // 50000*256*2 = 25.6 MB
    __hip_bfloat16* WT  = (__hip_bfloat16*)(ws + 25600000);             // 64 KB
    float* PRE = (float*)(ws + 25600000 + 65536);                       // 25.6 MB (also 'h')
    float* AGG = (float*)(ws + 25600000 + 65536 + 25600000);            // 25.6 MB
    float* RN  = (float*)(ws + 25600000 + 65536 + 2*25600000);          // 200 KB
    float* CS  = (float*)(ws + 25600000 + 65536 + 2*25600000 + 200000); // 512 B

    const __bf16* ABFr = (const __bf16*)ABF;
    const __bf16* WTr  = (const __bf16*)WT;

    const int T = 256;
    auto blocks = [](long long n, int t) { return (unsigned)((n + t - 1) / t); };

    // ---- input activations -> bf16 -------------------------------------
    f32_to_bf16<<<blocks((long long)M * 256, T), T, 0, stream>>>(in_feat, ABF, (long long)M * 256);

    // ===================== Layer 1: 256 -> 128 ==========================
    w_transpose_bf16<<<blocks(256 * 128, T), T, 0, stream>>>(W1, WT, 256, 128);
    gemm_bf16_norm<2><<<dim3((M + 15) / 16, 1), 128, 0, stream>>>(ABFr, WTr, norm, PRE, M, 256, 128);
    zero_f32<<<blocks((long long)M * 128, T), T, 0, stream>>>(AGG, (long long)M * 128);
    zero_f32<<<1, 128, 0, stream>>>(CS, 128);
    edge_scatter_add4<7><<<blocks((long long)E << 5, T), T, 0, stream>>>(PRE, src, dst, AGG, E);
    post_stats<128><<<256, 256, 0, stream>>>(AGG, norm, b1, PRE, RN, CS, M);
    finalize_relu_bf16<7><<<blocks((long long)M << 7, T), T, 0, stream>>>(PRE, RN, CS, ABF, M);

    // ===================== Layer 2: 128 -> 128 ==========================
    w_transpose_bf16<<<blocks(128 * 128, T), T, 0, stream>>>(W2, WT, 128, 128);
    gemm_bf16_norm<2><<<dim3((M + 15) / 16, 1), 128, 0, stream>>>(ABFr, WTr, norm, PRE, M, 128, 128);
    zero_f32<<<blocks((long long)M * 128, T), T, 0, stream>>>(AGG, (long long)M * 128);
    zero_f32<<<1, 128, 0, stream>>>(CS, 128);
    edge_scatter_add4<7><<<blocks((long long)E << 5, T), T, 0, stream>>>(PRE, src, dst, AGG, E);
    post_stats<128><<<256, 256, 0, stream>>>(AGG, norm, b2, PRE, RN, CS, M);
    finalize_relu_bf16<7><<<blocks((long long)M << 7, T), T, 0, stream>>>(PRE, RN, CS, ABF, M);

    // ===================== Layer 3: 128 -> 64 ===========================
    w_transpose_bf16<<<blocks(128 * 64, T), T, 0, stream>>>(W3, WT, 128, 64);
    gemm_bf16_norm<1><<<dim3((M + 15) / 16, 1), 128, 0, stream>>>(ABFr, WTr, norm, PRE, M, 128, 64);
    zero_f32<<<blocks((long long)M * 64, T), T, 0, stream>>>(AGG, (long long)M * 64);
    edge_scatter_add4<6><<<blocks((long long)E << 4, T), T, 0, stream>>>(PRE, src, dst, AGG, E);
    final_layer<6><<<blocks((long long)M << 6, T), T, 0, stream>>>(AGG, norm, b3, out, M);
}
